// MoEFFN_72198400246395
// MI455X (gfx1250) — matrix operand order repaired and checked by reading, not compiled, not run on hardware
//
#include <hip/hip_runtime.h>
#include <stdint.h>
#include <stddef.h>

// ---- problem sizes (from reference) ----
#define NTOK   8192            // B*T
#define RDIM   2048            // model dim
#define EDIM   1024            // expert hidden
#define NEXP   16
#define NTOPK  2
#define SHDIM  2048            // EXPERT_DIM * TOPK (shared hidden)
#define NSLOT  (NTOK * NTOPK)  // 16384 routed slots

// ---- GEMM tiling ----
#define BM 64
#define BN 128
#define BK 32
#define NTHREADS 256
#define PZ 8                   // persistent blocks per expert (m-direction)

typedef float        v4f   __attribute__((ext_vector_type(4)));
typedef float        v8f   __attribute__((ext_vector_type(8)));
typedef __bf16       v16bf __attribute__((ext_vector_type(16)));
typedef unsigned int v4u   __attribute__((ext_vector_type(4)));
typedef unsigned int v2u   __attribute__((ext_vector_type(2)));

union FragBF { v16bf bf; v4u q[2]; };

#define LDS_AS __attribute__((address_space(3)))

// pack two floats -> two bf16 (round-half-up) in one v_perm_b32
__device__ __forceinline__ unsigned pack_bf16(float lo, float hi) {
  union { float f; unsigned u; } a, b; a.f = lo; b.f = hi;
  return __builtin_amdgcn_perm(b.u + 0x8000u, a.u + 0x8000u, 0x07060302u);
}

__device__ __forceinline__ unsigned short f2bf(float f) {
  union { float f; unsigned u; } c; c.f = f;
  return (unsigned short)((c.u + 0x8000u) >> 16);
}

__device__ __forceinline__ float silu(float u) {
  return u * __builtin_amdgcn_rcpf(1.0f + __expf(-u));
}

// CDNA5 async tensor path: 16B global -> LDS, tracked by ASYNCcnt (no VGPR bounce)
__device__ __forceinline__ void async_cp16(unsigned short* lds_dst,
                                           const unsigned short* gsrc) {
  unsigned loff = (unsigned)(unsigned long long)(LDS_AS unsigned short*)lds_dst;
  asm volatile("global_load_async_to_lds_b128 %0, %1, off"
               :: "v"(loff), "v"((unsigned long long)gsrc)
               : "memory");
}
__device__ __forceinline__ void wait_async0() {
  asm volatile("s_wait_asynccnt 0x0" ::: "memory");
}

// =====================================================================
// GEMM1:  H[m, n] = silu( A[m,:] @ Wu[:,n] ) * ( A[m,:] @ Wg[:,n] )
// =====================================================================
__device__ __forceinline__ void gemm1_body(
    const unsigned short* __restrict__ xb,
    const float* __restrict__ Wu, const float* __restrict__ Wg, int ldb,
    unsigned short* __restrict__ Hout,
    const int* __restrict__ toks,          // slot->token, or nullptr (identity)
    int m0, int mcount, int n0,
    unsigned short (*As)[BK], unsigned short (*Bu)[BK], unsigned short (*Bg)[BK]) {

  const int tid  = threadIdx.x;
  const int lane = tid & 31;
  const int wave = tid >> 5;
  const int wm   = (wave >> 2) * 32;   // 0 / 32
  const int wn   = (wave & 3)  * 32;   // 0..96
  const int hl   = lane >> 4;          // half-wave
  const int ln   = lane & 15;

  v8f accU[2][2] = {{{0.f,0.f,0.f,0.f,0.f,0.f,0.f,0.f},{0.f,0.f,0.f,0.f,0.f,0.f,0.f,0.f}},
                    {{0.f,0.f,0.f,0.f,0.f,0.f,0.f,0.f},{0.f,0.f,0.f,0.f,0.f,0.f,0.f,0.f}}};
  v8f accG[2][2] = {{{0.f,0.f,0.f,0.f,0.f,0.f,0.f,0.f},{0.f,0.f,0.f,0.f,0.f,0.f,0.f,0.f}},
                    {{0.f,0.f,0.f,0.f,0.f,0.f,0.f,0.f},{0.f,0.f,0.f,0.f,0.f,0.f,0.f,0.f}}};

  for (int k0 = 0; k0 < RDIM; k0 += BK) {
    __syncthreads();
    { // A tile via async global->LDS copy (bf16, 16B per thread)
      const int r  = tid >> 2;
      const int c  = (tid & 3) * 8;
      const int sr = (r < mcount) ? (m0 + r) : m0;
      const int tok = toks ? toks[sr] : sr;
      async_cp16(&As[r][c], xb + (size_t)tok * RDIM + k0 + c);
    }
    { // B tiles (Wu, Wg): k-pair per thread, packed bf16 ds_store_b32
      const int kp = tid >> 4;            // 0..15 (k pair)
      const int nc = (tid & 15) * 8;      // 0..120
      const float* pu0 = Wu + (size_t)(k0 + 2 * kp) * ldb + n0 + nc;
      const float* pg0 = Wg + (size_t)(k0 + 2 * kp) * ldb + n0 + nc;
      if (k0 + BK < RDIM) {               // prefetch next K-slab of the weight stream
        __builtin_prefetch(pu0 + (size_t)BK * ldb, 0, 3);
        __builtin_prefetch(pg0 + (size_t)BK * ldb, 0, 3);
      }
      const v4f u0 = *(const v4f*)(pu0);      const v4f u0b = *(const v4f*)(pu0 + 4);
      const v4f u1 = *(const v4f*)(pu0 + ldb); const v4f u1b = *(const v4f*)(pu0 + ldb + 4);
      const v4f g0 = *(const v4f*)(pg0);      const v4f g0b = *(const v4f*)(pg0 + 4);
      const v4f g1 = *(const v4f*)(pg0 + ldb); const v4f g1b = *(const v4f*)(pg0 + ldb + 4);
#pragma unroll
      for (int i = 0; i < 4; ++i) {
        *(unsigned*)&Bu[nc + i][2 * kp]     = pack_bf16(u0[i],  u1[i]);
        *(unsigned*)&Bu[nc + 4 + i][2 * kp] = pack_bf16(u0b[i], u1b[i]);
        *(unsigned*)&Bg[nc + i][2 * kp]     = pack_bf16(g0[i],  g1[i]);
        *(unsigned*)&Bg[nc + 4 + i][2 * kp] = pack_bf16(g0b[i], g1b[i]);
      }
    }
    wait_async0();
    __syncthreads();

    // A fragments: lane = row (m%16); two 16B chunks per ISA 16-bit A layout
    FragBF a[2];
#pragma unroll
    for (int i = 0; i < 2; ++i) {
      const unsigned short* p = &As[wm + i * 16 + ln][hl * 8];
      a[i].q[0] = *(const v4u*)(p);
      a[i].q[1] = *(const v4u*)(p + 16);
    }
    // B fragments: lane = col, two 16B chunks (same K order as A)
    FragBF bu[2], bg[2];
#pragma unroll
    for (int j = 0; j < 2; ++j) {
      const unsigned short* pu2 = &Bu[wn + j * 16 + ln][hl * 8];
      bu[j].q[0] = *(const v4u*)(pu2);
      bu[j].q[1] = *(const v4u*)(pu2 + 16);
      const unsigned short* pg2 = &Bg[wn + j * 16 + ln][hl * 8];
      bg[j].q[0] = *(const v4u*)(pg2);
      bg[j].q[1] = *(const v4u*)(pg2 + 16);
    }
#pragma unroll
    for (int i = 0; i < 2; ++i)
#pragma unroll
      for (int j = 0; j < 2; ++j) {
        accU[i][j] = __builtin_amdgcn_wmma_f32_16x16x32_bf16(
            false, a[i].bf, false, bu[j].bf, (short)0, accU[i][j], false, false);
        accG[i][j] = __builtin_amdgcn_wmma_f32_16x16x32_bf16(
            false, a[i].bf, false, bg[j].bf, (short)0, accG[i][j], false, false);
      }
  }

  // epilogue: h = silu(u)*g, store bf16
#pragma unroll
  for (int i = 0; i < 2; ++i)
#pragma unroll
    for (int j = 0; j < 2; ++j)
#pragma unroll
      for (int r = 0; r < 8; ++r) {
        const int row = wm + i * 16 + hl * 8 + r;
        if (row < mcount) {
          const float h = silu(accU[i][j][r]) * accG[i][j][r];
          Hout[(size_t)(m0 + row) * ldb + (n0 + wn + j * 16 + ln)] = f2bf(h);
        }
      }
}

// =====================================================================
// GEMM2:  out[tok[m], n] (+)= w[m] * ( H[m,:] @ Wd[:,n] )
// =====================================================================
__device__ __forceinline__ void gemm2_body(
    const unsigned short* __restrict__ H, int kd,
    const float* __restrict__ Wd,          // [kd, RDIM] fp32
    float* __restrict__ out,
    const int* __restrict__ toks, const float* __restrict__ wts, // nullptr => identity/store
    int m0, int mcount, int n0,
    unsigned short (*As)[BK], unsigned short (*Bs)[BK],
    int* s_tok, float* s_w) {

  const int tid  = threadIdx.x;
  const int lane = tid & 31;
  const int wave = tid >> 5;
  const int wm   = (wave >> 2) * 32;
  const int wn   = (wave & 3)  * 32;
  const int hl   = lane >> 4;
  const int ln   = lane & 15;

  __syncthreads();                       // protect s_tok/s_w WAR across persistent iters
  if (toks && tid < BM) {
    if (tid < mcount) { s_tok[tid] = toks[m0 + tid]; s_w[tid] = wts[m0 + tid]; }
    else              { s_tok[tid] = 0;              s_w[tid] = 0.0f; }
  }

  v8f acc[2][2] = {{{0.f,0.f,0.f,0.f,0.f,0.f,0.f,0.f},{0.f,0.f,0.f,0.f,0.f,0.f,0.f,0.f}},
                   {{0.f,0.f,0.f,0.f,0.f,0.f,0.f,0.f},{0.f,0.f,0.f,0.f,0.f,0.f,0.f,0.f}}};

  for (int k0 = 0; k0 < kd; k0 += BK) {
    __syncthreads();
    { // A tile: async copy of bf16 hidden rows
      const int r  = tid >> 2;
      const int c  = (tid & 3) * 8;
      const int sr = (r < mcount) ? (m0 + r) : m0;
      async_cp16(&As[r][c], H + (size_t)sr * kd + k0 + c);
    }
    { // B tile (Wd fp32 -> packed bf16, transposed)
      const int kp = tid >> 4;
      const int nc = (tid & 15) * 8;
      const float* p0 = Wd + (size_t)(k0 + 2 * kp) * RDIM + n0 + nc;
      if (k0 + BK < kd) __builtin_prefetch(p0 + (size_t)BK * RDIM, 0, 3);
      const v4f w0 = *(const v4f*)(p0);       const v4f w0b = *(const v4f*)(p0 + 4);
      const v4f w1 = *(const v4f*)(p0 + RDIM); const v4f w1b = *(const v4f*)(p0 + RDIM + 4);
#pragma unroll
      for (int i = 0; i < 4; ++i) {
        *(unsigned*)&Bs[nc + i][2 * kp]     = pack_bf16(w0[i],  w1[i]);
        *(unsigned*)&Bs[nc + 4 + i][2 * kp] = pack_bf16(w0b[i], w1b[i]);
      }
    }
    wait_async0();
    __syncthreads();

    FragBF a[2], b[2];
#pragma unroll
    for (int i = 0; i < 2; ++i) {
      const unsigned short* p = &As[wm + i * 16 + ln][hl * 8];
      a[i].q[0] = *(const v4u*)(p);
      a[i].q[1] = *(const v4u*)(p + 16);
    }
#pragma unroll
    for (int j = 0; j < 2; ++j) {
      const unsigned short* p = &Bs[wn + j * 16 + ln][hl * 8];
      b[j].q[0] = *(const v4u*)(p);
      b[j].q[1] = *(const v4u*)(p + 16);
    }
#pragma unroll
    for (int i = 0; i < 2; ++i)
#pragma unroll
      for (int j = 0; j < 2; ++j)
        acc[i][j] = __builtin_amdgcn_wmma_f32_16x16x32_bf16(
            false, a[i].bf, false, b[j].bf, (short)0, acc[i][j], false, false);
  }

#pragma unroll
  for (int i = 0; i < 2; ++i)
#pragma unroll
    for (int j = 0; j < 2; ++j)
#pragma unroll
      for (int r = 0; r < 8; ++r) {
        const int row  = wm + i * 16 + hl * 8 + r;
        const int gcol = n0 + wn + j * 16 + ln;
        if (row < mcount) {
          if (toks) {
            atomicAdd(&out[(size_t)s_tok[row] * RDIM + gcol], s_w[row] * acc[i][j][r]);
          } else {
            out[(size_t)(m0 + row) * RDIM + gcol] = acc[i][j][r];
          }
        }
      }
}

// =====================================================================
// kernels
// =====================================================================
__global__ __launch_bounds__(NTHREADS) void cvt_x_k(const float* __restrict__ x,
                                                    unsigned short* __restrict__ xb) {
  const size_t i = ((size_t)blockIdx.x * blockDim.x + threadIdx.x) * 4;
  const v4f v = *(const v4f*)(x + i);
  v2u p;
  p[0] = pack_bf16(v[0], v[1]);
  p[1] = pack_bf16(v[2], v[3]);
  *(v2u*)(xb + i) = p;
}

__global__ __launch_bounds__(NTHREADS) void router_logits_k(
    const float* __restrict__ x, const float* __restrict__ rw, float* __restrict__ logits) {
  const int gid = blockIdx.x * blockDim.x + threadIdx.x;  // token*NEXP + expert
  const int n = gid >> 4, e = gid & 15;
  const v4f* xp = (const v4f*)(x + (size_t)n * RDIM);
  const v4f* wp = (const v4f*)(rw + (size_t)e * RDIM);
  float acc = 0.0f;
  for (int i = 0; i < RDIM / 4; ++i) {
    const v4f a = xp[i], b = wp[i];
    acc += a[0]*b[0] + a[1]*b[1] + a[2]*b[2] + a[3]*b[3];
  }
  logits[gid] = acc;
}

__global__ __launch_bounds__(NTHREADS) void router_topk_k(
    const float* __restrict__ logits, const float* __restrict__ bias,
    int* __restrict__ tki, float* __restrict__ tkw, int* __restrict__ counts) {
  const int n = blockIdx.x * blockDim.x + threadIdx.x;
  if (n >= NTOK) return;
  float l[NEXP], mx = -1e30f;
#pragma unroll
  for (int e = 0; e < NEXP; ++e) { l[e] = logits[n * NEXP + e]; mx = fmaxf(mx, l[e]); }
  float s[NEXP];
#pragma unroll
  for (int e = 0; e < NEXP; ++e) s[e] = __expf(l[e] - mx);
  int a1 = 0; float b1 = -1e30f;
#pragma unroll
  for (int e = 0; e < NEXP; ++e) { const float v = l[e] + bias[e]; if (v > b1) { b1 = v; a1 = e; } }
  int a2 = (a1 == 0) ? 1 : 0; float b2 = -1e30f;
#pragma unroll
  for (int e = 0; e < NEXP; ++e) if (e != a1) { const float v = l[e] + bias[e]; if (v > b2) { b2 = v; a2 = e; } }
  const float inv = __builtin_amdgcn_rcpf(s[a1] + s[a2]);
  tki[2 * n] = a1; tki[2 * n + 1] = a2;
  tkw[2 * n] = s[a1] * inv; tkw[2 * n + 1] = s[a2] * inv;
  atomicAdd(&counts[a1], 1);
  atomicAdd(&counts[a2], 1);
}

__global__ void init_counts_k(int* counts) {
  if (threadIdx.x < NEXP) counts[threadIdx.x] = 0;
}

__global__ void excl_scan_k(const int* __restrict__ counts, int* __restrict__ offs,
                            int* __restrict__ cursors) {
  if (threadIdx.x == 0) {
    int acc = 0;
    for (int e = 0; e < NEXP; ++e) { offs[e] = acc; cursors[e] = acc; acc += counts[e]; }
    offs[NEXP] = acc;
  }
}

__global__ __launch_bounds__(NTHREADS) void scatter_k(
    const int* __restrict__ tki, const float* __restrict__ tkw,
    int* __restrict__ cursors, int* __restrict__ btok, float* __restrict__ bw) {
  const int n = blockIdx.x * blockDim.x + threadIdx.x;
  if (n >= NTOK) return;
#pragma unroll
  for (int k = 0; k < NTOPK; ++k) {
    const int e = tki[2 * n + k];
    const int slot = atomicAdd(&cursors[e], 1);
    btok[slot] = n;
    bw[slot] = tkw[2 * n + k];
  }
}

__global__ __launch_bounds__(NTHREADS) void gemm1_shared_k(
    const unsigned short* __restrict__ xb, const float* __restrict__ Su,
    const float* __restrict__ Sg, unsigned short* __restrict__ Hs) {
  __shared__ __align__(16) unsigned short As[BM][BK];
  __shared__ __align__(16) unsigned short Bu[BN][BK];
  __shared__ __align__(16) unsigned short Bg[BN][BK];
  gemm1_body(xb, Su, Sg, SHDIM, Hs, nullptr,
             blockIdx.y * BM, BM, blockIdx.x * BN, As, Bu, Bg);
}

__global__ __launch_bounds__(NTHREADS) void gemm1_routed_k(
    const unsigned short* __restrict__ xb, const float* __restrict__ Wu,
    const float* __restrict__ Wg, unsigned short* __restrict__ Hr,
    const int* __restrict__ btok, const int* __restrict__ offs) {
  __shared__ __align__(16) unsigned short As[BM][BK];
  __shared__ __align__(16) unsigned short Bu[BN][BK];
  __shared__ __align__(16) unsigned short Bg[BN][BK];
  const int e = blockIdx.y;
  const float* wu = Wu + (size_t)e * RDIM * EDIM;
  const float* wg = Wg + (size_t)e * RDIM * EDIM;
  const int start = offs[e], end = offs[e + 1];
  const int n0 = blockIdx.x * BN;
  for (int m0 = start + blockIdx.z * BM; m0 < end; m0 += gridDim.z * BM) {
    int mc = end - m0; if (mc > BM) mc = BM;
    gemm1_body(xb, wu, wg, EDIM, Hr, btok, m0, mc, n0, As, Bu, Bg);
    __syncthreads();
  }
}

__global__ __launch_bounds__(NTHREADS) void gemm2_shared_k(
    const unsigned short* __restrict__ Hs, const float* __restrict__ Sd,
    float* __restrict__ out) {
  __shared__ __align__(16) unsigned short As[BM][BK];
  __shared__ __align__(16) unsigned short Bs[BN][BK];
  __shared__ int s_tok[BM];
  __shared__ float s_w[BM];
  gemm2_body(Hs, SHDIM, Sd, out, nullptr, nullptr,
             blockIdx.y * BM, BM, blockIdx.x * BN, As, Bs, s_tok, s_w);
}

__global__ __launch_bounds__(NTHREADS) void gemm2_routed_k(
    const unsigned short* __restrict__ Hr, const float* __restrict__ Wd,
    float* __restrict__ out, const int* __restrict__ btok,
    const float* __restrict__ bw, const int* __restrict__ offs) {
  __shared__ __align__(16) unsigned short As[BM][BK];
  __shared__ __align__(16) unsigned short Bs[BN][BK];
  __shared__ int s_tok[BM];
  __shared__ float s_w[BM];
  const int e = blockIdx.y;
  const float* wd = Wd + (size_t)e * EDIM * RDIM;
  const int start = offs[e], end = offs[e + 1];
  const int n0 = blockIdx.x * BN;
  for (int m0 = start + blockIdx.z * BM; m0 < end; m0 += gridDim.z * BM) {
    int mc = end - m0; if (mc > BM) mc = BM;
    gemm2_body(Hr, EDIM, wd, out, btok, bw, m0, mc, n0, As, Bs, s_tok, s_w);
    __syncthreads();
  }
}

// =====================================================================
extern "C" void kernel_launch(void* const* d_in, const int* in_sizes, int n_in,
                              void* d_out, int out_size, void* d_ws, size_t ws_size,
                              hipStream_t stream) {
  const float* x  = (const float*)d_in[0];
  const float* rw = (const float*)d_in[1];
  const float* rb = (const float*)d_in[2];
  const float* Wu = (const float*)d_in[3];
  const float* Wg = (const float*)d_in[4];
  const float* Wd = (const float*)d_in[5];
  const float* Su = (const float*)d_in[6];
  const float* Sg = (const float*)d_in[7];
  const float* Sd = (const float*)d_in[8];
  float* out = (float*)d_out;

  char* w = (char*)d_ws;
  size_t off = 0;
  auto bump = [&](size_t bytes) { void* p = w + off; off = (off + bytes + 255) & ~(size_t)255; return p; };

  unsigned short* xb = (unsigned short*)bump((size_t)NTOK * RDIM * 2);     // 32 MB
  unsigned short* Hs = (unsigned short*)bump((size_t)NTOK * SHDIM * 2);    // 32 MB
  unsigned short* Hr = Hs;  // alias: Hs dead before Hr is written (stream-ordered)
  float* logits = (float*)bump((size_t)NTOK * NEXP * 4);
  int*   tki    = (int*)bump((size_t)NTOK * NTOPK * 4);
  float* tkw    = (float*)bump((size_t)NTOK * NTOPK * 4);
  int*   btok   = (int*)bump((size_t)(NSLOT + BM) * 4);
  float* bw     = (float*)bump((size_t)(NSLOT + BM) * 4);
  int*   counts = (int*)bump(64 * 4);
  int*   offs   = (int*)bump(64 * 4);
  int*   curs   = (int*)bump(64 * 4);

  // routing
  init_counts_k<<<1, 32, 0, stream>>>(counts);
  cvt_x_k<<<((size_t)NTOK * RDIM / 4) / NTHREADS, NTHREADS, 0, stream>>>(x, xb);
  router_logits_k<<<(NTOK * NEXP) / NTHREADS, NTHREADS, 0, stream>>>(x, rw, logits);
  router_topk_k<<<NTOK / NTHREADS, NTHREADS, 0, stream>>>(logits, rb, tki, tkw, counts);
  excl_scan_k<<<1, 1, 0, stream>>>(counts, offs, curs);
  scatter_k<<<NTOK / NTHREADS, NTHREADS, 0, stream>>>(tki, tkw, curs, btok, bw);

  // shared expert (writes every element of out -> initializes d_out)
  gemm1_shared_k<<<dim3(SHDIM / BN, NTOK / BM), NTHREADS, 0, stream>>>(xb, Su, Sg, Hs);
  gemm2_shared_k<<<dim3(RDIM / BN, NTOK / BM), NTHREADS, 0, stream>>>(Hs, Sd, out);

  // routed experts (atomic accumulate into out)
  gemm1_routed_k<<<dim3(EDIM / BN, NEXP, PZ), NTHREADS, 0, stream>>>(xb, Wu, Wg, Hr, btok, offs);
  gemm2_routed_k<<<dim3(RDIM / BN, NEXP, PZ), NTHREADS, 0, stream>>>(Hr, Wd, out, btok, bw, offs);
}
